// NonLocalBlock_19602230739585
// MI455X (gfx1250) — matrix operand order repaired and checked by reading, not compiled, not run on hardware
//
#include <hip/hip_runtime.h>

// ---------------- problem constants ----------------
#define BB 4
#define CC 256
#define CI 128
#define NN 4096   // H*W = 64*64

typedef __attribute__((ext_vector_type(16))) __bf16 v16bf;
typedef __attribute__((ext_vector_type(8)))  float  v8f;
typedef __attribute__((ext_vector_type(8)))  int    v8i;
typedef __attribute__((ext_vector_type(4)))  int    v4i;

#if __has_builtin(__builtin_amdgcn_global_load_async_to_lds_b128) && \
    __has_builtin(__builtin_amdgcn_s_wait_asynccnt)
#define USE_ASYNC 1
#else
#define USE_ASYNC 0
#endif

#define GLOBAL_AS __attribute__((address_space(1)))
#define LDS_AS    __attribute__((address_space(3)))

__device__ __forceinline__ unsigned short f32_to_bf16(float f) {
    __bf16 h = (__bf16)f;                 // native RNE convert
    return __builtin_bit_cast(unsigned short, h);
}

// A-fragment: 16x32 bf16 (M x K), src row-major [16][>=32], ld in elements.
// ISA 7.12.2: lanes 0-15 row M=lane, K {0..7,16..23}; lanes 16-31 same rows, K {8..15,24..31}.
__device__ __forceinline__ v16bf load_frag_a(const unsigned short* src, int ld) {
    const int lane = threadIdx.x & 31;
    const int m  = lane & 15;
    const int kb = (lane >> 4) << 3;          // 0 or 8
    const unsigned short* p = src + m * ld + kb;
    v8i r;
#pragma unroll
    for (int v = 0; v < 8; ++v) {
        const int k = ((v & 4) << 2) + ((v & 3) << 1); // 0,2,4,6,16,18,20,22
        r[v] = *reinterpret_cast<const int*>(p + k);   // bf16 pair (K, K+1)
    }
    return __builtin_bit_cast(v16bf, r);
}

// B-fragment: 32x16 bf16 (K x N), loaded from the TRANSPOSED source BT[n][k]
// (row-major, K contiguous). lanes 0-15: col n, K {0..7,16..23}; lanes 16-31: col n, K {8..15,24..31}.
__device__ __forceinline__ v16bf load_frag_b(const unsigned short* srcT, int ld) {
    const int lane = threadIdx.x & 31;
    const unsigned short* p = srcT + (lane & 15) * ld + ((lane >> 4) << 3);
    const int4 lo = *reinterpret_cast<const int4*>(p);
    const int4 hi = *reinterpret_cast<const int4*>(p + 16);
    v8i r;
    r[0] = lo.x; r[1] = lo.y; r[2] = lo.z; r[3] = lo.w;
    r[4] = hi.x; r[5] = hi.y; r[6] = hi.z; r[7] = hi.w;
    return __builtin_bit_cast(v16bf, r);
}

__device__ __forceinline__ v8f wmma_bf16(v16bf a, v16bf b, v8f c) {
    return __builtin_amdgcn_wmma_f32_16x16x32_bf16(false, a, false, b,
                                                   (short)0, c, false, false);
}

// 16-byte global -> LDS copy (async DMA when available)
__device__ __forceinline__ void cp16(unsigned short* lds, const unsigned short* g) {
#if USE_ASYNC
    __builtin_amdgcn_global_load_async_to_lds_b128(
        (GLOBAL_AS v4i*)g,
        (LDS_AS v4i*)lds, 0, 0);
#else
    *reinterpret_cast<int4*>(lds) = *reinterpret_cast<const int4*>(g);
#endif
}

// ---------------- K0: fp32 -> bf16 weights (theta, phi, g) ----------------
__global__ void k_cvt_w(const float* tw, const float* pw, const float* gw,
                        unsigned short* WB) {
    const int sz = CI * CC;
    int idx = blockIdx.x * blockDim.x + threadIdx.x;
    if (idx >= 3 * sz) return;
    const float* src = (idx < sz) ? tw : (idx < 2 * sz) ? pw : gw;
    WB[idx] = f32_to_bf16(src[idx % sz]);
}

// ---------------- K1: x [B][C][N] fp32 -> XT [B][N][C] bf16 ----------------
__global__ void k_transpose(const float* x, unsigned short* XT) {
    __shared__ float tile[32][33];
    const int b  = blockIdx.z;
    const int n0 = blockIdx.x * 32;
    const int c0 = blockIdx.y * 32;
    const int tx = threadIdx.x, ty = threadIdx.y;     // 32 x 8
#pragma unroll
    for (int r = 0; r < 4; ++r) {
        int cc = c0 + ty + r * 8;
        tile[ty + r * 8][tx] = x[((size_t)(b * CC + cc) * NN) + n0 + tx];
    }
    __syncthreads();
#pragma unroll
    for (int r = 0; r < 4; ++r) {
        int nn = n0 + ty + r * 8;
        XT[((size_t)(b * NN + nn) * CC) + c0 + tx] = f32_to_bf16(tile[tx][ty + r * 8]);
    }
}

// ---------------- K2: projections via WMMA ----------------
// OutT[n][ci] = sum_c XT[n][c] * w[ci][c] + bias[ci]
// theta/phi stored [B][N][CI] bf16; g stored [B][CI][N] fp32.
__global__ void k_proj(const unsigned short* XT, const unsigned short* WB,
                       const float* tb, const float* pb, const float* gb,
                       unsigned short* THETA, unsigned short* PHI, float* GRAW) {
    const int b    = blockIdx.y;
    const int wid  = threadIdx.x >> 5;
    const int lane = threadIdx.x & 31;
    const int n0   = (blockIdx.x * 4 + wid) * 16;
    const int col   = lane & 15;          // D-tile column (= ci within tile)
    const int rbase = (lane >> 4) << 3;   // D-tile row base (= n within tile)

    const unsigned short* xrow = XT + (size_t)(b * NN + n0) * CC;
    v16bf a[8];
#pragma unroll
    for (int ks = 0; ks < 8; ++ks) a[ks] = load_frag_a(xrow + ks * 32, CC);

    for (int p = 0; p < 3; ++p) {
        const float* bias = (p == 0) ? tb : (p == 1) ? pb : gb;
        for (int ct = 0; ct < 8; ++ct) {
            const int ci = ct * 16 + col;
            const float bv = bias[ci];
            v8f acc;
#pragma unroll
            for (int r = 0; r < 8; ++r) acc[r] = bv;
            const unsigned short* wrow = WB + ((size_t)p * CI + ct * 16) * CC;
#pragma unroll
            for (int ks = 0; ks < 8; ++ks)
                acc = wmma_bf16(a[ks], load_frag_b(wrow + ks * 32, CC), acc);
            if (p < 2) {
                unsigned short* dst = (p == 0) ? THETA : PHI;
#pragma unroll
                for (int r = 0; r < 8; ++r)
                    dst[(size_t)(b * NN + n0 + rbase + r) * CI + ci] = f32_to_bf16(acc[r]);
            } else {
                float* gd = GRAW + ((size_t)(b * CI + ci) * NN) + n0 + rbase;
#pragma unroll
                for (int r = 0; r < 8; ++r) gd[r] = acc[r];
            }
        }
    }
}

// ---------------- K3: per-column softmax stats (m[j], Z[j]) ----------------
// f[i,j] = sum_c THETA[i][c]*PHI[j][c]; softmax over i for each j.
__global__ void k_stats(const unsigned short* THETA, const unsigned short* PHI,
                        float* MOUT, float* ZOUT) {
    const int b    = blockIdx.y;
    const int j0   = blockIdx.x * 16;
    const int wid  = threadIdx.x >> 5;
    const int lane = threadIdx.x & 31;
    __shared__ float sm[4][16], sz[4][16];

    v16bf bfrag[4];
    const unsigned short* prow = PHI + (size_t)(b * NN + j0) * CI;
#pragma unroll
    for (int ks = 0; ks < 4; ++ks) bfrag[ks] = load_frag_b(prow + ks * 32, CI);

    float m = -1e30f, z = 0.f;
    for (int it = wid; it < NN / 16; it += 4) {
        const unsigned short* trow = THETA + (size_t)(b * NN + it * 16) * CI;
        v8f f;
#pragma unroll
        for (int r = 0; r < 8; ++r) f[r] = 0.f;
#pragma unroll
        for (int ks = 0; ks < 4; ++ks)
            f = wmma_bf16(load_frag_a(trow + ks * 32, CI), bfrag[ks], f);
        float tm = f[0];
#pragma unroll
        for (int r = 1; r < 8; ++r) tm = fmaxf(tm, f[r]);
        const float Mn = fmaxf(m, tm);
        float s = 0.f;
#pragma unroll
        for (int r = 0; r < 8; ++r) s += __expf(f[r] - Mn);
        z = z * __expf(m - Mn) + s;
        m = Mn;
    }
    { // lanes L and L+16 hold the two row-halves of the same column j
        const float m2 = __shfl_xor(m, 16, 32);
        const float z2 = __shfl_xor(z, 16, 32);
        const float Mn = fmaxf(m, m2);
        z = z * __expf(m - Mn) + z2 * __expf(m2 - Mn);
        m = Mn;
    }
    if (lane < 16) { sm[wid][lane] = m; sz[wid][lane] = z; }
    __syncthreads();
    if (threadIdx.x < 16) {
        float mm = sm[0][threadIdx.x], zz = sz[0][threadIdx.x];
#pragma unroll
        for (int w = 1; w < 4; ++w) {
            const float mw = sm[w][threadIdx.x], zw = sz[w][threadIdx.x];
            const float Mx = fmaxf(mm, mw);
            zz = zz * __expf(mm - Mx) + zw * __expf(mw - Mx);
            mm = Mx;
        }
        MOUT[b * NN + j0 + threadIdx.x] = mm;
        ZOUT[b * NN + j0 + threadIdx.x] = zz;
    }
}

// ---------------- K4: fold 1/Z into g, convert to bf16 ----------------
__global__ void k_scale_g(const float* GRAW, const float* Zb, unsigned short* GHAT) {
    int idx = blockIdx.x * blockDim.x + threadIdx.x;
    if (idx >= BB * CI * NN) return;
    const int j = idx & (NN - 1);
    const int b = idx >> 19;                  // CI*NN = 2^19
    GHAT[idx] = f32_to_bf16(GRAW[idx] / Zb[b * NN + j]);
}

// ---------------- K5: flash pass -> Y[b][i][ci] ----------------
// y[i,c] = sum_j exp(f[i,j]-m[j]) * GHAT[c][j]
// Per block: 4 waves, each one 16-row i-tile. phi(32x128) and ghat(128x32)
// j-tiles are staged cooperatively in LDS (async DMA, double buffered).
__global__ void k_attn(const unsigned short* THETA, const unsigned short* PHI,
                       const unsigned short* GHAT, const float* Mv, float* Y) {
    const int b    = blockIdx.y;
    const int wid  = threadIdx.x >> 5;
    const int lane = threadIdx.x & 31;
    const int tid  = threadIdx.x;            // 0..127
    const int i0   = (blockIdx.x * 4 + wid) * 16;

    __shared__ unsigned short sPhi[2][32 * CI];    // [j][c]
    __shared__ unsigned short sGhat[2][CI * 32];   // [c][j]
    __shared__ unsigned short ebuf[4][16 * 34];
    unsigned short* eb = ebuf[wid];

    const int col   = lane & 15;
    const int rbase = (lane >> 4) << 3;

    v16bf afrag[4];
    const unsigned short* trow = THETA + (size_t)(b * NN + i0) * CI;
#pragma unroll
    for (int ks = 0; ks < 4; ++ks) afrag[ks] = load_frag_a(trow + ks * 32, CI);

    v8f yacc[8];
#pragma unroll
    for (int c8 = 0; c8 < 8; ++c8)
#pragma unroll
        for (int r = 0; r < 8; ++r) yacc[c8][r] = 0.f;

    const unsigned short* phiBase = PHI + (size_t)b * NN * CI;
    const unsigned short* gBase   = GHAT + (size_t)b * CI * NN;
    const float* mBase            = Mv + b * NN;

    // stage tile for jb into buffer s: 8 async instructions per wave
    auto stage = [&](int jb, int s) {
        const unsigned short* pg = phiBase + (size_t)jb * CI;   // 8KB contiguous
#pragma unroll
        for (int q = 0; q < 4; ++q) {
            const int off = (tid + q * 128) * 8;                // 16B chunks
            cp16(&sPhi[s][off], pg + off);
        }
        const unsigned short* gg = gBase + (size_t)tid * NN + jb; // row ci=tid, 64B
#pragma unroll
        for (int q = 0; q < 4; ++q)
            cp16(&sGhat[s][tid * 32 + q * 8], gg + q * 8);
    };

    const int NT = NN / 32;  // 128 j-tiles
    stage(0, 0);
    for (int t = 0; t < NT; ++t) {
        const int cur = t & 1;
        if (t + 1 < NT) {
            stage((t + 1) * 32, cur ^ 1);
#if USE_ASYNC
            __builtin_amdgcn_s_wait_asynccnt(8);   // current tile's 8 done, next 8 in flight
#endif
        } else {
#if USE_ASYNC
            __builtin_amdgcn_s_wait_asynccnt(0);
#endif
        }
        __syncthreads();                           // tiles visible to all waves

        const int jb = t * 32;
#pragma unroll
        for (int h = 0; h < 2; ++h) {
            v8f f;
#pragma unroll
            for (int r = 0; r < 8; ++r) f[r] = 0.f;
#pragma unroll
            for (int ks = 0; ks < 4; ++ks)
                f = wmma_bf16(afrag[ks],
                              load_frag_b(&sPhi[cur][(h * 16) * CI + ks * 32], CI), f);
            const float mj = mBase[jb + h * 16 + col];
#pragma unroll
            for (int r = 0; r < 8; ++r)
                eb[(rbase + r) * 34 + h * 16 + col] = f32_to_bf16(__expf(f[r] - mj));
        }
        const v16bf ea = load_frag_a(eb, 34);      // same-wave LDS RAW: hw/compiler waits
#pragma unroll
        for (int c8 = 0; c8 < 8; ++c8) {
            const v16bf bg = load_frag_b(&sGhat[cur][(c8 * 16) * 32], 32);
            yacc[c8] = wmma_bf16(ea, bg, yacc[c8]);
        }
        __syncthreads();                           // done reading before next overwrite
    }
#pragma unroll
    for (int c8 = 0; c8 < 8; ++c8) {
        const int ci = c8 * 16 + col;
#pragma unroll
        for (int r = 0; r < 8; ++r)
            Y[(size_t)(b * NN + i0 + rbase + r) * CI + ci] = yacc[c8][r];
    }
}

// ---------------- K6: wz 1x1 conv (fp32 VALU, LDS-tiled) ----------------
__global__ void k_wz(const float* Y, const float* wzw, const float* wzb, float* WY) {
    const int b  = blockIdx.y;
    const int n0 = blockIdx.x * 16;
    __shared__ float sY[16][CI];
    for (int idx = threadIdx.x; idx < 16 * CI; idx += blockDim.x) {
        const int nn = idx >> 7, ci = idx & (CI - 1);
        sY[nn][ci] = Y[(size_t)(b * NN + n0 + nn) * CI + ci];
    }
    __syncthreads();
    const int co = threadIdx.x;               // blockDim.x == 256 == CC
    float acc[16];
    const float bv = wzb[co];
#pragma unroll
    for (int nn = 0; nn < 16; ++nn) acc[nn] = bv;
    const float* wr = wzw + (size_t)co * CI;
    for (int ci = 0; ci < CI; ++ci) {
        const float w = wr[ci];
#pragma unroll
        for (int nn = 0; nn < 16; ++nn) acc[nn] += w * sY[nn][ci];
    }
    float* dst = WY + ((size_t)(b * CC + co) * NN) + n0;
#pragma unroll
    for (int nn = 0; nn < 16; ++nn) dst[nn] = acc[nn];
}

// ---------------- K7: BN batch statistics per channel ----------------
__global__ void k_bn_stats(const float* WY, float* stats) {
    const int co = blockIdx.x;
    __shared__ float ss[256], sq[256];
    float s = 0.f, q = 0.f;
    for (int idx = threadIdx.x; idx < BB * NN; idx += 256) {
        const int bb = idx >> 12, nn = idx & (NN - 1);
        const float v = WY[((size_t)(bb * CC + co) * NN) + nn];
        s += v; q += v * v;
    }
    ss[threadIdx.x] = s; sq[threadIdx.x] = q;
    __syncthreads();
    for (int off = 128; off > 0; off >>= 1) {
        if ((int)threadIdx.x < off) {
            ss[threadIdx.x] += ss[threadIdx.x + off];
            sq[threadIdx.x] += sq[threadIdx.x + off];
        }
        __syncthreads();
    }
    if (threadIdx.x == 0) {
        const float inv = 1.f / (float)(BB * NN);
        const float mean = ss[0] * inv;
        const float var  = sq[0] * inv - mean * mean;
        stats[co]      = mean;
        stats[CC + co] = rsqrtf(var + 1e-5f);
    }
}

// ---------------- K8: normalize + affine + residual ----------------
__global__ void k_bn_apply(const float* WY, const float* x, const float* stats,
                           const float* gamma, const float* beta, float* out) {
    int idx = blockIdx.x * blockDim.x + threadIdx.x;
    if (idx >= BB * CC * NN) return;
    const int c = (idx >> 12) & (CC - 1);
    out[idx] = (WY[idx] - stats[c]) * stats[CC + c] * gamma[c] + beta[c] + x[idx];
}

// ---------------- launch ----------------
extern "C" void kernel_launch(void* const* d_in, const int* in_sizes, int n_in,
                              void* d_out, int out_size, void* d_ws, size_t ws_size,
                              hipStream_t stream) {
    (void)in_sizes; (void)n_in; (void)out_size; (void)ws_size;
    const float* x       = (const float*)d_in[0];
    const float* g_w     = (const float*)d_in[1];
    const float* g_b     = (const float*)d_in[2];
    const float* theta_w = (const float*)d_in[3];
    const float* theta_b = (const float*)d_in[4];
    const float* phi_w   = (const float*)d_in[5];
    const float* phi_b   = (const float*)d_in[6];
    const float* wz_w    = (const float*)d_in[7];
    const float* wz_b    = (const float*)d_in[8];
    const float* gamma   = (const float*)d_in[9];
    const float* beta    = (const float*)d_in[10];
    float* out = (float*)d_out;

    char* ws = (char*)d_ws;
    size_t off = 0;
    auto alloc = [&](size_t bytes) -> void* {
        void* p = ws + off;
        off += (bytes + 255) & ~(size_t)255;
        return p;
    };
    unsigned short* XT    = (unsigned short*)alloc((size_t)BB * NN * CC * 2);
    unsigned short* WB    = (unsigned short*)alloc((size_t)3 * CI * CC * 2);
    unsigned short* THETA = (unsigned short*)alloc((size_t)BB * NN * CI * 2);
    unsigned short* PHI   = (unsigned short*)alloc((size_t)BB * NN * CI * 2);
    float*          GRAW  = (float*)alloc((size_t)BB * CI * NN * 4);
    unsigned short* GHAT  = (unsigned short*)alloc((size_t)BB * CI * NN * 2);
    float*          MB    = (float*)alloc((size_t)BB * NN * 4);
    float*          ZB    = (float*)alloc((size_t)BB * NN * 4);
    float*          Ybuf  = (float*)alloc((size_t)BB * NN * CI * 4);
    float*          WY    = (float*)alloc((size_t)BB * CC * NN * 4);
    float*          STATS = (float*)alloc((size_t)2 * CC * 4);

    k_cvt_w<<<(3 * CI * CC + 255) / 256, 256, 0, stream>>>(theta_w, phi_w, g_w, WB);
    k_transpose<<<dim3(NN / 32, CC / 32, BB), dim3(32, 8), 0, stream>>>(x, XT);
    k_proj<<<dim3(NN / 64, BB), 128, 0, stream>>>(XT, WB, theta_b, phi_b, g_b,
                                                  THETA, PHI, GRAW);
    k_stats<<<dim3(NN / 16, BB), 128, 0, stream>>>(THETA, PHI, MB, ZB);
    k_scale_g<<<(BB * CI * NN + 255) / 256, 256, 0, stream>>>(GRAW, ZB, GHAT);
    k_attn<<<dim3(NN / 64, BB), 128, 0, stream>>>(THETA, PHI, GHAT, MB, Ybuf);
    k_wz<<<dim3(NN / 16, BB), 256, 0, stream>>>(Ybuf, wz_w, wz_b, WY);
    k_bn_stats<<<CC, 256, 0, stream>>>(WY, STATS);
    k_bn_apply<<<(BB * CC * NN + 255) / 256, 256, 0, stream>>>(WY, x, STATS,
                                                               gamma, beta, out);
}
